// Attention_76639396429963
// MI455X (gfx1250) — matrix operand order repaired and checked by reading, not compiled, not run on hardware
//
#include <hip/hip_runtime.h>
#include <hip/hip_bf16.h>

// ---------------- problem constants (from reference) ----------------
#define BSZ    2
#define SQL    2048
#define DIMSZ  4096
#define NH     32
#define NKV    8
#define HD     128
#define KVDIM  (NKV * HD)      // 1024
#define TOKENS (BSZ * SQL)     // 4096

typedef __attribute__((ext_vector_type(16))) __bf16 v16bf;
typedef __attribute__((ext_vector_type(8)))  __bf16 v8bf;
typedef __attribute__((ext_vector_type(4)))  __bf16 v4bf;
typedef __attribute__((ext_vector_type(8)))  float  v8f;

// D = A(16x32 bf16) * B(32x16 bf16) + C(16x16 f32), wave32 WMMA
__device__ __forceinline__ v8f wmma_bf16(v16bf a, v16bf b, v8f c) {
  return __builtin_amdgcn_wmma_f32_16x16x32_bf16(false, a, false, b, (short)0, c,
                                                 false, false);
}

// A-matrix fragment (16x32, 16-bit): lane L<16 holds row L, k {0..7,16..23};
// lane L+16 holds row L, k {8..15,24..31}.  p -> row-major [row][k], ld elems.
__device__ __forceinline__ v16bf load_fragA(const __bf16* p, int ld) {
  const int lane = threadIdx.x & 31;
  const int r    = lane & 15;
  const int kk   = (lane >> 4) << 3;        // 0 or 8
  const __bf16* q = p + r * ld + kk;
  v8bf lo = *(const v8bf*)q;                // k = kk .. kk+7
  v8bf hi = *(const v8bf*)(q + 16);         // k = kk+16 .. kk+23
  v16bf f;
#pragma unroll
  for (int i = 0; i < 8; ++i) { f[i] = lo[i]; f[i + 8] = hi[i]; }
  return f;
}

// B-matrix fragment (32x16, 16-bit): lane N<16 holds col N, k {0..7,16..23};
// lane N+16 holds col N, k {8..15,24..31}.  p -> row-major [n][k] tile, ld elems.
__device__ __forceinline__ v16bf load_fragB(const __bf16* p, int ld) {
  const int lane = threadIdx.x & 31;
  const int r    = lane & 15;
  const int kk   = (lane >> 4) << 3;        // 0 or 8
  const __bf16* q = p + r * ld + kk;
  v8bf lo = *(const v8bf*)q;                // k = kk .. kk+7
  v8bf hi = *(const v8bf*)(q + 16);         // k = kk+16 .. kk+23
  v16bf f;
#pragma unroll
  for (int i = 0; i < 8; ++i) { f[i] = lo[i]; f[i + 8] = hi[i]; }
  return f;
}

// Stage a 128x64 tile into LDS (stride 72 bf16), converting f32->bf16 if needed.
// f32 path packs 4 converted values into one 8B DS store (ds_store_b64).
template <typename T>
__device__ __forceinline__ void stage128x64(__bf16* dst, const T* src, int ld) {
  const int tid = threadIdx.x;              // 256 threads
  if constexpr (sizeof(T) == 4) {
#pragma unroll
    for (int i = 0; i < 8; ++i) {
      int c = tid + i * 256;                // 2048 chunks of 4 floats
      int row = c >> 4, col = (c & 15) << 2;
      float4 v = *(const float4*)(src + (size_t)row * ld + col);
      v4bf pk = { (__bf16)v.x, (__bf16)v.y, (__bf16)v.z, (__bf16)v.w };
      *(v4bf*)(dst + row * 72 + col) = pk;
    }
  } else {
#pragma unroll
    for (int i = 0; i < 4; ++i) {
      int c = tid + i * 256;                // 1024 chunks of 8 bf16
      int row = c >> 3, col = (c & 7) << 3;
      *(v8bf*)(dst + row * 72 + col) = *(const v8bf*)(src + (size_t)row * ld + col);
    }
  }
}

// ---------------- GEMM: C[M,N] = A[M,K] * B[N,K]^T ----------------
// 256 threads = 8 waves; block tile 128x128, k-step 64 (2 WMMA steps per stage);
// wave tile 64x32 (4x2 frags) -> 16 WMMAs per wave between barriers.
template <typename TA, typename TB, typename TC>
__global__ __launch_bounds__(256) void gemm_wmma(const TA* __restrict__ A,
                                                 const TB* __restrict__ B,
                                                 TC* __restrict__ C,
                                                 int M, int N, int K) {
  __shared__ __bf16 As[128 * 72];
  __shared__ __bf16 Bs[128 * 72];
  const int tid = threadIdx.x;
  const int wid = tid >> 5;
  const int wm = wid >> 2;                  // 0..1
  const int wn = wid & 3;                   // 0..3
  const int m0 = blockIdx.y * 128;
  const int n0 = blockIdx.x * 128;
  v8f acc[4][2] = {};
  for (int k0 = 0; k0 < K; k0 += 64) {
    // prefetch next K-tile while this one is staged/computed (global_prefetch_b8)
    if (k0 + 64 < K) {
      int prow = tid >> 1, pcol = (tid & 1) << 5;
      __builtin_prefetch(A + (size_t)(m0 + prow) * K + k0 + 64 + pcol, 0, 3);
      __builtin_prefetch(B + (size_t)(n0 + prow) * K + k0 + 64 + pcol, 0, 3);
    }
    stage128x64(As, A + (size_t)m0 * K + k0, K);
    stage128x64(Bs, B + (size_t)n0 * K + k0, K);
    __syncthreads();
#pragma unroll
    for (int ks = 0; ks < 2; ++ks) {
      v16bf af[4], bf2[2];
#pragma unroll
      for (int mi = 0; mi < 4; ++mi)
        af[mi] = load_fragA(&As[(wm * 64 + mi * 16) * 72 + ks * 32], 72);
#pragma unroll
      for (int ni = 0; ni < 2; ++ni)
        bf2[ni] = load_fragB(&Bs[(wn * 32 + ni * 16) * 72 + ks * 32], 72);
#pragma unroll
      for (int mi = 0; mi < 4; ++mi)
#pragma unroll
        for (int ni = 0; ni < 2; ++ni)
          acc[mi][ni] = wmma_bf16(af[mi], bf2[ni], acc[mi][ni]);
    }
    __syncthreads();
  }
  // C layout: lane = col (lane&15); rows = vgpr e + 8*(lane>=16)
  const int lane = tid & 31, cn = lane & 15, half = lane >> 4;
#pragma unroll
  for (int mi = 0; mi < 4; ++mi)
#pragma unroll
    for (int ni = 0; ni < 2; ++ni)
#pragma unroll
      for (int e = 0; e < 8; ++e) {
        int row = m0 + wm * 64 + mi * 16 + e + 8 * half;
        int col = n0 + wn * 32 + ni * 16 + cn;
        C[(size_t)row * N + col] = (TC)acc[mi][ni][e];
      }
}

// ---------------- RoPE, in place on bf16 [token, ncols] ----------------
__global__ __launch_bounds__(256) void rope_kernel(__bf16* __restrict__ t, int ncols,
                                                   const float* __restrict__ cosT,
                                                   const float* __restrict__ sinT,
                                                   int total_pairs) {
  int p = blockIdx.x * 256 + threadIdx.x;
  if (p >= total_pairs) return;
  int halfc = ncols >> 1;
  int tok = p / halfc;
  int w   = p - tok * halfc;
  int head = w >> 6;                        // / (HD/2)
  int i    = w & 63;
  int s    = tok & (SQL - 1);
  size_t idx = (size_t)tok * ncols + head * HD + 2 * i;
  float a = (float)t[idx], b = (float)t[idx + 1];
  float c = cosT[s * 64 + i], si = sinT[s * 64 + i];
  t[idx]     = (__bf16)(a * c - b * si);
  t[idx + 1] = (__bf16)(a * si + b * c);
}

// ---------------- Flash attention ----------------
// grid (SQL/64, NH, BSZ), 128 threads = 4 waves; wave owns 16 query rows.
__global__ __launch_bounds__(128) void attn_kernel(const __bf16* __restrict__ Q,
                                                   const __bf16* __restrict__ Kc,
                                                   const __bf16* __restrict__ Vc,
                                                   __bf16* __restrict__ O) {
  constexpr int LDK = 136;   // 32 kv rows x 128 d  (272B rows, 16B aligned)
  constexpr int LDV = 40;    // 128 d rows x 32 kv  (80B rows)
  constexpr int LDP = 40;    // 16 rows x 32 cols per wave
  __shared__ __bf16 Ks[32 * LDK];
  __shared__ __bf16 VT[128 * LDV];
  __shared__ __bf16 Ps[4][16 * LDP];

  const int tid = threadIdx.x;
  const int w = tid >> 5, lane = tid & 31, cn = lane & 15, half = lane >> 4;
  const int h = blockIdx.y, b = blockIdx.z;
  const int m0 = blockIdx.x * 64;
  const int kvh = h >> 2;                   // GQA: 4 q-heads per kv-head
  const int mq = m0 + w * 16;               // first seq row of this wave
  const int tok0 = b * SQL + mq;

  // Q fragments: 4 k-steps of 32 over head_dim=128 (A layout)
  v16bf qf[4];
  const __bf16* qp = Q + (size_t)tok0 * DIMSZ + h * HD;
#pragma unroll
  for (int ds = 0; ds < 4; ++ds) qf[ds] = load_fragA(qp + ds * 32, DIMSZ);

  float mrun[8], lrun[8];
  v8f o[8] = {};
#pragma unroll
  for (int e = 0; e < 8; ++e) { mrun[e] = -1e30f; lrun[e] = 0.0f; }
  const float scale = 0.08838834764831845f; // 1/sqrt(128)

  const int kv_end = m0 + 64;               // causal: last tile covers row m0+63
  for (int t0 = 0; t0 < kv_end; t0 += 32) {
    __syncthreads();
    // stage K tile [32 kv][128 d] and V^T tile [128 d][32 kv]
#pragma unroll
    for (int i = 0; i < 4; ++i) {
      int c = tid + i * 128;                // 512 chunks of 8 bf16
      int t = c >> 4, d0 = (c & 15) << 3;
      size_t g = (size_t)(b * SQL + t0 + t) * KVDIM + kvh * HD + d0;
      *(v8bf*)&Ks[t * LDK + d0] = *(const v8bf*)(Kc + g);
      v8bf v = *(const v8bf*)(Vc + g);
#pragma unroll
      for (int j = 0; j < 8; ++j) VT[(d0 + j) * LDV + t] = v[j];
    }
    __syncthreads();

    // S = Q K^T : two 16x16 score frags (kv cols t0..t0+15 and t0+16..t0+31)
    v8f s0 = {}, s1 = {};
#pragma unroll
    for (int ds = 0; ds < 4; ++ds) {
      s0 = wmma_bf16(qf[ds], load_fragB(&Ks[0 * LDK + ds * 32], LDK), s0);
      s1 = wmma_bf16(qf[ds], load_fragB(&Ks[16 * LDK + ds * 32], LDK), s1);
    }

    // online softmax (row = e + 8*half within wave tile; col = lane&15)
    float p0[8], p1[8], fs[8];
#pragma unroll
    for (int e = 0; e < 8; ++e) {
      int row = mq + e + 8 * half;          // absolute query position
      float v0 = s0[e] * scale, v1 = s1[e] * scale;
      if (t0 + cn > row)      v0 = -1e9f;   // causal mask
      if (t0 + 16 + cn > row) v1 = -1e9f;
      float mx = fmaxf(v0, v1);
      mx = fmaxf(mx, __shfl_xor(mx, 8, 32));
      mx = fmaxf(mx, __shfl_xor(mx, 4, 32));
      mx = fmaxf(mx, __shfl_xor(mx, 2, 32));
      mx = fmaxf(mx, __shfl_xor(mx, 1, 32));
      float mnew = fmaxf(mrun[e], mx);
      float f = __expf(mrun[e] - mnew);
      float e0 = __expf(v0 - mnew);
      float e1 = __expf(v1 - mnew);
      float sum = e0 + e1;
      sum += __shfl_xor(sum, 8, 32);
      sum += __shfl_xor(sum, 4, 32);
      sum += __shfl_xor(sum, 2, 32);
      sum += __shfl_xor(sum, 1, 32);
      lrun[e] = lrun[e] * f + sum;
      mrun[e] = mnew;
      fs[e] = f; p0[e] = e0; p1[e] = e1;
    }
#pragma unroll
    for (int nf = 0; nf < 8; ++nf)
#pragma unroll
      for (int e = 0; e < 8; ++e) o[nf][e] *= fs[e];

    // round-trip P through per-wave LDS to get the A-fragment layout
#pragma unroll
    for (int e = 0; e < 8; ++e) {
      Ps[w][(e + 8 * half) * LDP + cn]      = (__bf16)p0[e];
      Ps[w][(e + 8 * half) * LDP + 16 + cn] = (__bf16)p1[e];
    }
    v16bf pf = load_fragA(&Ps[w][0], LDP);

    // O += P V : V^T rows are output dims, cols are kv positions (B layout)
#pragma unroll
    for (int nf = 0; nf < 8; ++nf)
      o[nf] = wmma_bf16(pf, load_fragB(&VT[(nf * 16) * LDV], LDV), o[nf]);
  }

  // normalize + write bf16 output [token, h*128 + d]
#pragma unroll
  for (int nf = 0; nf < 8; ++nf)
#pragma unroll
    for (int e = 0; e < 8; ++e) {
      int tok = tok0 + e + 8 * half;
      int col = h * HD + nf * 16 + cn;
      O[(size_t)tok * DIMSZ + col] = (__bf16)(o[nf][e] / lrun[e]);
    }
}

// ---------------- host launcher ----------------
extern "C" void kernel_launch(void* const* d_in, const int* in_sizes, int n_in,
                              void* d_out, int out_size, void* d_ws, size_t ws_size,
                              hipStream_t stream) {
  (void)in_sizes; (void)n_in; (void)out_size; (void)ws_size;
  const float* x  = (const float*)d_in[0];
  const float* wq = (const float*)d_in[1];
  const float* wk = (const float*)d_in[2];
  const float* wv = (const float*)d_in[3];
  const float* wo = (const float*)d_in[4];
  // d_in[5..6] caches unused (start_pos==0); d_in[7..8] rope tables; [9] mask; [10] start_pos
  const float* fcos = (const float*)d_in[7];
  const float* fsin = (const float*)d_in[8];
  float* out = (float*)d_out;

  // workspace layout (bf16): q_pre | k_pre | v | attn  (~80 MB total)
  __bf16* q_pre = (__bf16*)d_ws;                                  // TOKENS x 4096
  __bf16* k_pre = q_pre + (size_t)TOKENS * DIMSZ;                 // TOKENS x 1024
  __bf16* v_b   = k_pre + (size_t)TOKENS * KVDIM;                 // TOKENS x 1024
  __bf16* attn  = v_b   + (size_t)TOKENS * KVDIM;                 // TOKENS x 4096

  // 1) QKV projections (f32 -> bf16 WMMA -> bf16)
  gemm_wmma<float, float, __bf16><<<dim3(DIMSZ / 128, TOKENS / 128), 256, 0, stream>>>(
      x, wq, q_pre, TOKENS, DIMSZ, DIMSZ);
  gemm_wmma<float, float, __bf16><<<dim3(KVDIM / 128, TOKENS / 128), 256, 0, stream>>>(
      x, wk, k_pre, TOKENS, KVDIM, DIMSZ);
  gemm_wmma<float, float, __bf16><<<dim3(KVDIM / 128, TOKENS / 128), 256, 0, stream>>>(
      x, wv, v_b, TOKENS, KVDIM, DIMSZ);

  // 2) RoPE in place on q and k (q_pre/k_pre are rewritten by GEMMs each call)
  {
    int qpairs = TOKENS * (DIMSZ / 2);
    int kpairs = TOKENS * (KVDIM / 2);
    rope_kernel<<<(qpairs + 255) / 256, 256, 0, stream>>>(q_pre, DIMSZ, fcos, fsin, qpairs);
    rope_kernel<<<(kpairs + 255) / 256, 256, 0, stream>>>(k_pre, KVDIM, fcos, fsin, kpairs);
  }

  // 3) causal flash attention with GQA
  attn_kernel<<<dim3(SQL / 64, NH, BSZ), 128, 0, stream>>>(q_pre, k_pre, v_b, attn);

  // 4) output projection (bf16 x (f32->bf16) -> f32)
  gemm_wmma<__bf16, float, float><<<dim3(DIMSZ / 128, TOKENS / 128), 256, 0, stream>>>(
      attn, wo, out, TOKENS, DIMSZ, DIMSZ);
}
